// CAM_Module_42365557408071
// MI455X (gfx1250) — compile-verified, matrix operand order checked by reading, not run on hardware
//
#include <hip/hip_runtime.h>

// CAM module: B=16, C=512, HW=4096
//   energy = v v^T (bf16 WMMA, f32 acc, symmetric: upper tile-pairs only),
//   softmax(-energy) rows, out = att @ v, d_out = gamma*out + x.
// Workspace layout (92.3 MB total, assumed <= ws_size):
//   [0)              bf16 v      : 16*512*4096 * 2 = 67,108,864 B
//   [67108864)       f32  energy : 16*512*512  * 4 = 16,777,216 B
//   [83886080)       bf16 att    : 16*512*512  * 2 =  8,388,608 B

#define BATCH 16
#define CCH   512
#define HWN   4096

typedef __attribute__((ext_vector_type(16))) __bf16 v16bf;
typedef __attribute__((ext_vector_type(8)))  __bf16 v8bf;
typedef __attribute__((ext_vector_type(8)))  float  v8f;
typedef unsigned short u16;

static __device__ __forceinline__ u16 f32_to_bf16(float f) {
  unsigned int u = __float_as_uint(f);
  u += 0x7FFFu + ((u >> 16) & 1u);          // round-to-nearest-even
  return (u16)(u >> 16);
}

// Per-lane fragment for 16-bit 16x32 A (and the Gram-symmetric B):
// lane half h: elems 0..7 <- K = h*8 .. h*8+7 ; elems 8..15 <- K = 16+h*8 ..
// Both chunks are contiguous 16-byte loads from one row.
static __device__ __forceinline__ v16bf load_frag_row(const __bf16* p, int hv) {
  union { v16bf v; v8bf h[2]; } f;
  f.h[0] = *(const v8bf*)(p + hv * 8);
  f.h[1] = *(const v8bf*)(p + 16 + hv * 8);
  return f.v;
}

static __device__ __forceinline__ v8f wmma_bf16(v16bf a, v16bf b, v8f c) {
  return __builtin_amdgcn_wmma_f32_16x16x32_bf16(false, a, false, b,
                                                 (short)0, c, false, false);
}

// C/D layout: VGPR r, lanes 0-15: M=r, N=lane; lanes 16-31: M=8+r, N=lane-16
static __device__ __forceinline__ void store_tile(float* e, int bm0, int n0,
                                                  v8f acc, int hv, int lr) {
  float* p = e + (size_t)bm0 * CCH + n0;
#pragma unroll
  for (int r = 0; r < 8; ++r)
    p[(size_t)(r + 8 * hv) * CCH + lr] = acc[r];
}

// Mirrored (transposed) store: acc[m][n] -> e[n0+n][m0+m] (4B scatter, L2-resident)
static __device__ __forceinline__ void store_tile_T(float* e, int bn0, int m0,
                                                    v8f acc, int hv, int lr) {
  float* p = e + (size_t)bn0 * CCH + m0 + 8 * hv;
#pragma unroll
  for (int r = 0; r < 8; ++r)
    p[(size_t)lr * CCH + r] = acc[r];
}

// ---------------- kernel 1: f32 -> bf16 convert ----------------
__global__ void cam_cvt_bf16(const float4* __restrict__ x,
                             unsigned long long* __restrict__ out, long n4) {
  long i = (long)blockIdx.x * blockDim.x + threadIdx.x;
  if (i >= n4) return;
  float4 f = x[i];
  unsigned long long r = (unsigned long long)f32_to_bf16(f.x)
                       | ((unsigned long long)f32_to_bf16(f.y) << 16)
                       | ((unsigned long long)f32_to_bf16(f.z) << 32)
                       | ((unsigned long long)f32_to_bf16(f.w) << 48);
  out[i] = r;
}

// ---------------- kernel 2: energy = v v^T (symmetric Gram) ----------------
// grid = (C/32/8, C/32, B), block = 256 (8 waves).
// One wave computes a 32x32 block (2x2 WMMA tiles) at (cPair, dPair);
// only dPair >= cPair is computed, off-diagonal blocks mirror-stored.
__global__ void cam_gram(const u16* __restrict__ vbf, float* __restrict__ energy) {
  const int b     = blockIdx.z;
  const int cPair = blockIdx.y;                       // 0..15 (M block / 32)
  const int w     = threadIdx.x >> 5;
  const int dPair = blockIdx.x * 8 + w;               // 0..15 (N block / 32)
  if (dPair < cPair) return;                          // symmetry
  const int lane = threadIdx.x & 31;
  const int hv   = lane >> 4;
  const int lr   = lane & 15;

  const __bf16* vb = (const __bf16*)vbf + (size_t)b * CCH * HWN;
  const __bf16* a0 = vb + (size_t)(cPair * 32 + lr) * HWN;
  const __bf16* a1 = a0 + (size_t)16 * HWN;
  const __bf16* b0 = vb + (size_t)(dPair * 32 + lr) * HWN;
  const __bf16* b1 = b0 + (size_t)16 * HWN;

  v8f acc00 = {}, acc01 = {}, acc10 = {}, acc11 = {};
  for (int k0 = 0; k0 < HWN; k0 += 32) {
    __builtin_prefetch(a0 + k0 + 512, 0, 1);          // global_prefetch_b8
    __builtin_prefetch(a1 + k0 + 512, 0, 1);
    __builtin_prefetch(b0 + k0 + 512, 0, 1);
    __builtin_prefetch(b1 + k0 + 512, 0, 1);
    v16bf fa0 = load_frag_row(a0 + k0, hv);
    v16bf fa1 = load_frag_row(a1 + k0, hv);
    v16bf fb0 = load_frag_row(b0 + k0, hv);
    v16bf fb1 = load_frag_row(b1 + k0, hv);
    acc00 = wmma_bf16(fa0, fb0, acc00);
    acc01 = wmma_bf16(fa0, fb1, acc01);
    acc10 = wmma_bf16(fa1, fb0, acc10);
    acc11 = wmma_bf16(fa1, fb1, acc11);
  }

  float* eb = energy + (size_t)b * CCH * CCH;
  const int m0 = cPair * 32, n0 = dPair * 32;
  store_tile(eb, m0,      n0,      acc00, hv, lr);
  store_tile(eb, m0,      n0 + 16, acc01, hv, lr);
  store_tile(eb, m0 + 16, n0,      acc10, hv, lr);
  store_tile(eb, m0 + 16, n0 + 16, acc11, hv, lr);
  if (dPair > cPair) {                                // mirror below diagonal
    store_tile_T(eb, n0,      m0,      acc00, hv, lr);
    store_tile_T(eb, n0 + 16, m0,      acc01, hv, lr);
    store_tile_T(eb, n0,      m0 + 16, acc10, hv, lr);
    store_tile_T(eb, n0 + 16, m0 + 16, acc11, hv, lr);
  }
}

// ---------------- kernel 3: softmax rows -> bf16 attention ----------------
// softmax(max_row - e) stabilized == exp(min_row - e_j) / sum.
// One wave per 512-element row; grid = 8192/8 blocks of 256.
__global__ void cam_softmax(const float* __restrict__ energy, u16* __restrict__ attbf) {
  const int row  = blockIdx.x * 8 + (threadIdx.x >> 5);   // 0..8191
  const int lane = threadIdx.x & 31;
  const float* e = energy + (size_t)row * CCH;

  float v[16];
  float mn = 3.4e38f;
#pragma unroll
  for (int i = 0; i < 16; ++i) {
    v[i] = e[lane + 32 * i];
    mn = fminf(mn, v[i]);
  }
#pragma unroll
  for (int off = 16; off > 0; off >>= 1)
    mn = fminf(mn, __shfl_xor(mn, off, 32));

  float s = 0.f;
#pragma unroll
  for (int i = 0; i < 16; ++i) {
    v[i] = __expf(mn - v[i]);
    s += v[i];
  }
#pragma unroll
  for (int off = 16; off > 0; off >>= 1)
    s += __shfl_xor(s, off, 32);

  const float inv = __frcp_rn(s);
  u16* arow = attbf + (size_t)row * CCH;
#pragma unroll
  for (int i = 0; i < 16; ++i)
    arow[lane + 32 * i] = f32_to_bf16(v[i] * inv);
}

// ---------------- kernel 4: out = att @ v, d_out = gamma*out + x ----------------
// grid = (HW/16/8, C/64, B), block = 256 (8 waves).
// One wave computes 4 c-tiles x 1 n-tile: the expensive gathered B fragment
// (columns of v) is loaded once and amortized over 4 WMMAs.
__global__ void cam_av(const u16* __restrict__ attbf, const u16* __restrict__ vbf,
                       const float* __restrict__ x, const float* __restrict__ gammaP,
                       float* __restrict__ out) {
  const int b     = blockIdx.z;
  const int cQuad = blockIdx.y;                       // 0..7  (M block / 64)
  const int w     = threadIdx.x >> 5;
  const int nTile = blockIdx.x * 8 + w;               // 0..255 (N tile)
  const int lane  = threadIdx.x & 31;
  const int hv    = lane >> 4;
  const int lr    = lane & 15;

  const __bf16* att0 = (const __bf16*)attbf + (size_t)b * CCH * CCH
                     + (size_t)(cQuad * 64 + lr) * CCH;
  const __bf16* att1 = att0 + (size_t)16 * CCH;
  const __bf16* att2 = att0 + (size_t)32 * CCH;
  const __bf16* att3 = att0 + (size_t)48 * CCH;
  const __bf16* vb   = (const __bf16*)vbf + (size_t)b * CCH * HWN;
  const int ncol     = nTile * 16 + lr;

  v8f acc0 = {}, acc1 = {}, acc2 = {}, acc3 = {};
  for (int k0 = 0; k0 < CCH; k0 += 32) {
    // Shared B fragment: B[k][n] = v[k0+k][ncol] (u16 gathers, coalesced over lr)
    union { v16bf v; __bf16 e[16]; } bf;
#pragma unroll
    for (int t = 0; t < 16; ++t) {
      int k = (t < 8) ? (hv * 8 + t) : (16 + hv * 8 + (t - 8));
      bf.e[t] = vb[(size_t)(k0 + k) * HWN + ncol];
    }
    v16bf fa0 = load_frag_row(att0 + k0, hv);
    v16bf fa1 = load_frag_row(att1 + k0, hv);
    v16bf fa2 = load_frag_row(att2 + k0, hv);
    v16bf fa3 = load_frag_row(att3 + k0, hv);
    acc0 = wmma_bf16(fa0, bf.v, acc0);
    acc1 = wmma_bf16(fa1, bf.v, acc1);
    acc2 = wmma_bf16(fa2, bf.v, acc2);
    acc3 = wmma_bf16(fa3, bf.v, acc3);
  }

  const float g = gammaP[0];
  const size_t base = ((size_t)b * CCH + (size_t)cQuad * 64) * HWN + (size_t)nTile * 16;
#pragma unroll
  for (int r = 0; r < 8; ++r) {
    int m = r + 8 * hv;
    size_t i0 = base + (size_t)m * HWN + lr;
    size_t i1 = i0 + (size_t)16 * HWN;
    size_t i2 = i0 + (size_t)32 * HWN;
    size_t i3 = i0 + (size_t)48 * HWN;
    out[i0] = g * acc0[r] + x[i0];
    out[i1] = g * acc1[r] + x[i1];
    out[i2] = g * acc2[r] + x[i2];
    out[i3] = g * acc3[r] + x[i3];
  }
}

// ---------------- host launcher ----------------
extern "C" void kernel_launch(void* const* d_in, const int* in_sizes, int n_in,
                              void* d_out, int out_size, void* d_ws, size_t ws_size,
                              hipStream_t stream) {
  (void)in_sizes; (void)n_in; (void)out_size; (void)ws_size;

  const float* x     = (const float*)d_in[0];
  const float* gamma = (const float*)d_in[1];

  const size_t VBF_BYTES = (size_t)BATCH * CCH * HWN * 2;      // 67,108,864
  const size_t EN_BYTES  = (size_t)BATCH * CCH * CCH * 4;      // 16,777,216

  u16*   vbf    = (u16*)d_ws;
  float* energy = (float*)((char*)d_ws + VBF_BYTES);
  u16*   attbf  = (u16*)((char*)d_ws + VBF_BYTES + EN_BYTES);

  // 1) convert x -> bf16 (float4 per thread)
  {
    long n4 = (long)BATCH * CCH * HWN / 4;                     // 8,388,608
    int blocks = (int)((n4 + 255) / 256);
    cam_cvt_bf16<<<blocks, 256, 0, stream>>>((const float4*)x,
                                             (unsigned long long*)vbf, n4);
  }
  // 2) energy = v v^T (upper tile-pairs + mirror)
  cam_gram<<<dim3(CCH / 32 / 8, CCH / 32, BATCH), 256, 0, stream>>>(vbf, energy);
  // 3) softmax rows -> bf16 attention
  cam_softmax<<<dim3(BATCH * CCH / 8), 256, 0, stream>>>(energy, attbf);
  // 4) out = gamma*(att@v) + x
  cam_av<<<dim3(HWN / 16 / 8, CCH / 64, BATCH), 256, 0, stream>>>(attbf, vbf, x,
                                                                  gamma, (float*)d_out);
}
